// GroupedQueryAttention_59459527246149
// MI455X (gfx1250) — matrix operand order repaired and checked by reading, not compiled, not run on hardware
//
#include <hip/hip_runtime.h>
#include <hip/hip_bf16.h>

// GQA: B=4, N=2048, QH=8, KVH=4, DH=64, E=512, KVE=256, G=2, softmax_one, causal.

#define DEVINL __device__ __forceinline__

typedef __attribute__((ext_vector_type(16))) __bf16 v16bf;
typedef __attribute__((ext_vector_type(8)))  float  v8f;
typedef __attribute__((ext_vector_type(4)))  unsigned int u32x4;

union FragAB {
  v16bf v;
  unsigned short us[16];
  u32x4 q[2];
};

DEVINL unsigned short f2bf(float f) {
  unsigned int u = __float_as_uint(f);
  u += 0x7fffu + ((u >> 16) & 1u);   // round-to-nearest-even
  return (unsigned short)(u >> 16);
}

DEVINL v8f zero8() { v8f z = {0.f,0.f,0.f,0.f,0.f,0.f,0.f,0.f}; return z; }

DEVINL v8f wmma_bf16(const FragAB& a, const FragAB& b, v8f c) {
  // (neg_a, A, neg_b, B, c_mod, C, reuse_a, reuse_b)
  return __builtin_amdgcn_wmma_f32_16x16x32_bf16(false, a.v, false, b.v,
                                                 (short)0, c, false, false);
}

// ---------------------------------------------------------------------------
// Weight convert: W (512 x Nc, f32 row-major) -> Wt (Nc x 512, bf16) so that
// WMMA B-fragments are per-lane contiguous 32B loads.
__global__ void wconv_kernel(const float* __restrict__ W,
                             unsigned short* __restrict__ Wt,
                             int Nc, int total) {
  int i = blockIdx.x * blockDim.x + threadIdx.x;
  if (i >= total) return;
  int k = i / Nc, n = i % Nc;
  Wt[(size_t)n * 512 + k] = f2bf(W[i]);
}

// ---------------------------------------------------------------------------
// LayerNorm over last dim (512), f32 in -> bf16 out. One block per row.
__global__ __launch_bounds__(256) void ln_kernel(const float* __restrict__ x,
                                                 const float* __restrict__ g,
                                                 const float* __restrict__ bta,
                                                 unsigned short* __restrict__ out) {
  const int row = blockIdx.x;
  const int tid = threadIdx.x;
  const float* xr = x + (size_t)row * 512;
  float a0 = xr[tid], a1 = xr[tid + 256];
  __shared__ float sh[256], sh2[256];
  sh[tid]  = a0 + a1;
  sh2[tid] = a0 * a0 + a1 * a1;
  __syncthreads();
  for (int off = 128; off > 0; off >>= 1) {
    if (tid < off) { sh[tid] += sh[tid + off]; sh2[tid] += sh2[tid + off]; }
    __syncthreads();
  }
  const float mean = sh[0] * (1.f / 512.f);
  const float var  = sh2[0] * (1.f / 512.f) - mean * mean;
  const float rstd = rsqrtf(var + 1e-5f);
  unsigned short* o = out + (size_t)row * 512;
  o[tid]       = f2bf((a0 - mean) * rstd * g[tid]       + bta[tid]);
  o[tid + 256] = f2bf((a1 - mean) * rstd * g[tid + 256] + bta[tid + 256]);
}

// ---------------------------------------------------------------------------
// GEMM: C[M x Nc] = A[M x 512](bf16) @ Wt[Nc x 512]^T (bf16) + bias, via WMMA.
// Wave tile 64x64 (4x4 16x16 frags -> 16 WMMAs per 8 fragment loads per K-step).
// Block 256 thr = 8 waves (4M x 2N) -> block tile 256x128.
// mode 0: f32 row-major out. mode 1: bf16 head-major [b][h][tok][64].
// mode 3: bf16 transposed   [b][h][64][tok]  (for V).
__global__ __launch_bounds__(256) void gemm_bf16_kernel(
    const unsigned short* __restrict__ A,
    const unsigned short* __restrict__ Bt,
    const float* __restrict__ bias,
    void* __restrict__ outp,
    int Nc, int mode, int H) {
  const int lane = threadIdx.x & 31;
  const int wave = threadIdx.x >> 5;
  const int l16  = lane & 15;
  const int half = lane >> 4;
  const int row0 = blockIdx.x * 256 + (wave >> 1) * 64;
  const int col0 = blockIdx.y * 128 + (wave & 1) * 64;

  v8f acc[4][4];
  #pragma unroll
  for (int i = 0; i < 4; i++)
    #pragma unroll
    for (int j = 0; j < 4; j++) acc[i][j] = zero8();

  for (int k0 = 0; k0 < 512; k0 += 32) {
    FragAB a[4], b[4];
    #pragma unroll
    for (int i = 0; i < 4; i++) {
      const unsigned short* p = A + (size_t)(row0 + i * 16 + l16) * 512 + k0;
      a[i].q[0] = *(const u32x4*)(p + half * 16);
      a[i].q[1] = *(const u32x4*)(p + half * 16 + 8);
    }
    #pragma unroll
    for (int j = 0; j < 4; j++) {
      const unsigned short* p = Bt + (size_t)(col0 + j * 16 + l16) * 512 + k0 + half * 16;
      b[j].q[0] = *(const u32x4*)(p);
      b[j].q[1] = *(const u32x4*)(p + 8);
    }
    #pragma unroll
    for (int i = 0; i < 4; i++)
      #pragma unroll
      for (int j = 0; j < 4; j++)
        acc[i][j] = wmma_bf16(a[i], b[j], acc[i][j]);
  }

  #pragma unroll
  for (int i = 0; i < 4; i++) {
    #pragma unroll
    for (int j = 0; j < 4; j++) {
      #pragma unroll
      for (int r = 0; r < 8; r++) {
        const int row = row0 + i * 16 + r + half * 8;
        const int col = col0 + j * 16 + l16;
        const float val = acc[i][j][r] + bias[col];
        if (mode == 0) {
          ((float*)outp)[(size_t)row * Nc + col] = val;
        } else {
          const int bb  = row >> 11;      // / 2048
          const int tok = row & 2047;
          const int h   = col >> 6;
          const int d   = col & 63;
          if (mode == 1) {
            ((unsigned short*)outp)[(((size_t)bb * H + h) * 2048 + tok) * 64 + d] = f2bf(val);
          } else {  // mode 3: transposed V
            ((unsigned short*)outp)[(((size_t)bb * H + h) * 64 + d) * 2048 + tok] = f2bf(val);
          }
        }
      }
    }
  }
}

// ---------------------------------------------------------------------------
// Flash-style causal attention with softmax_one. One wave per (b, qh, 32-row
// q tile): two 16-row subtiles share each K/V fragment (16 WMMAs / 8 loads
// per 32-key step). softmax_one == online softmax with phantom logit 0.
__global__ __launch_bounds__(32) void attn_kernel(
    const unsigned short* __restrict__ Qb,   // [B][8][2048][64] bf16
    const unsigned short* __restrict__ Kb,   // [B][4][2048][64] bf16
    const unsigned short* __restrict__ Vt,   // [B][4][64][2048] bf16
    float* __restrict__ attn_out) {          // [B][2048][512] f32
  __shared__ __align__(16) unsigned short Plds[32 * 32];
  const int lane = threadIdx.x & 31;
  const int l16  = lane & 15;
  const int half = lane >> 4;
  const int tile = blockIdx.x & 63;          // N/32 = 64 tiles
  const int qh   = (blockIdx.x >> 6) & 7;
  const int b    = blockIdx.x >> 9;
  const int kvh  = qh >> 1;                  // G = 2
  const int m0   = tile * 32;
  const float scale = 0.125f;                // 1/sqrt(64)

  // Resident Q A-fragments: u = row-subtile (0,1), i = d-half (0..31, 32..63)
  FragAB qa[2][2];
  #pragma unroll
  for (int u = 0; u < 2; u++) {
    const unsigned short* qrow =
        Qb + (((size_t)b * 8 + qh) * 2048 + (m0 + u * 16 + l16)) * 64;
    #pragma unroll
    for (int i = 0; i < 2; i++) {
      const unsigned short* p = qrow + i * 32;
      qa[u][i].q[0] = *(const u32x4*)(p + half * 16);
      qa[u][i].q[1] = *(const u32x4*)(p + half * 16 + 8);
    }
  }
  const unsigned short* Kh = Kb + ((size_t)b * 4 + kvh) * 2048 * 64;
  const unsigned short* Vh = Vt + ((size_t)b * 4 + kvh) * 64 * 2048;

  float mrun[2][8], lrun[2][8];
  v8f acc[2][4];
  #pragma unroll
  for (int u = 0; u < 2; u++) {
    #pragma unroll
    for (int r = 0; r < 8; r++) { mrun[u][r] = 0.f; lrun[u][r] = 1.f; }
    #pragma unroll
    for (int j = 0; j < 4; j++) acc[u][j] = zero8();
  }

  const int nst = tile + 1;                  // causal: keys 0 .. m0+31
  for (int t = 0; t < nst; ++t) {
    const int s0 = t * 32;
    // ---- load K fragments once, reuse for both row-subtiles ----
    FragAB kf[2][2];                         // [sub (key 16-group)][d-half]
    #pragma unroll
    for (int sub = 0; sub < 2; ++sub)
      #pragma unroll
      for (int i = 0; i < 2; i++) {
        const unsigned short* p =
            Kh + (size_t)(s0 + sub * 16 + l16) * 64 + i * 32 + half * 16;
        kf[sub][i].q[0] = *(const u32x4*)(p);
        kf[sub][i].q[1] = *(const u32x4*)(p + 8);
      }
    // ---- S = Q @ K^T : 2 subtiles x 2 key-groups x 2 d-halves = 8 WMMAs ----
    v8f sf[2][2];
    #pragma unroll
    for (int u = 0; u < 2; u++)
      #pragma unroll
      for (int sub = 0; sub < 2; ++sub) {
        sf[u][sub] = zero8();
        #pragma unroll
        for (int i = 0; i < 2; i++)
          sf[u][sub] = wmma_bf16(qa[u][i], kf[sub][i], sf[u][sub]);
      }
    // ---- mask + scale, online softmax_one update, stage P to LDS ----
    #pragma unroll
    for (int u = 0; u < 2; u++) {
      #pragma unroll
      for (int r = 0; r < 8; r++) {
        const int row = m0 + u * 16 + r + half * 8;
        #pragma unroll
        for (int sub = 0; sub < 2; ++sub) {
          const int col = s0 + sub * 16 + l16;
          float v = sf[u][sub][r] * scale;
          if (col > row) v = -1e30f;
          sf[u][sub][r] = v;
        }
        float m = fmaxf(sf[u][0][r], sf[u][1][r]);
        m = fmaxf(m, __shfl_xor(m, 1, 32));
        m = fmaxf(m, __shfl_xor(m, 2, 32));
        m = fmaxf(m, __shfl_xor(m, 4, 32));
        m = fmaxf(m, __shfl_xor(m, 8, 32));
        const float mnew = fmaxf(mrun[u][r], m);
        const float corr = __expf(mrun[u][r] - mnew);
        const float p0 = __expf(sf[u][0][r] - mnew);
        const float p1 = __expf(sf[u][1][r] - mnew);
        float rs = p0 + p1;
        rs += __shfl_xor(rs, 1, 32);
        rs += __shfl_xor(rs, 2, 32);
        rs += __shfl_xor(rs, 4, 32);
        rs += __shfl_xor(rs, 8, 32);
        lrun[u][r] = lrun[u][r] * corr + rs;
        mrun[u][r] = mnew;
        #pragma unroll
        for (int j = 0; j < 4; j++) acc[u][j][r] *= corr;
        const int prow = u * 16 + r + half * 8;
        Plds[prow * 32 + l16]      = f2bf(p0);
        Plds[prow * 32 + 16 + l16] = f2bf(p1);
      }
    }
    __syncthreads();
    // ---- re-gather P as A-fragments (k = key index) ----
    FragAB pa[2];
    #pragma unroll
    for (int u = 0; u < 2; u++) {
      const unsigned short* p = &Plds[(u * 16 + l16) * 32];
      pa[u].q[0] = *(const u32x4*)(p + half * 16);
      pa[u].q[1] = *(const u32x4*)(p + half * 16 + 8);
    }
    __syncthreads();
    // ---- acc += P @ V : V fragments shared by both subtiles ----
    #pragma unroll
    for (int j = 0; j < 4; j++) {
      FragAB vf;
      const unsigned short* p = Vh + (size_t)(j * 16 + l16) * 2048 + s0 + half * 16;
      vf.q[0] = *(const u32x4*)(p);
      vf.q[1] = *(const u32x4*)(p + 8);
      #pragma unroll
      for (int u = 0; u < 2; u++)
        acc[u][j] = wmma_bf16(pa[u], vf, acc[u][j]);
    }
  }

  // ---- write O / l : attn_out[b][tok][qh*64 + d] ----
  #pragma unroll
  for (int u = 0; u < 2; u++)
    #pragma unroll
    for (int j = 0; j < 4; j++)
      #pragma unroll
      for (int r = 0; r < 8; r++) {
        const int tok = m0 + u * 16 + r + half * 8;
        const int col = qh * 64 + j * 16 + l16;
        attn_out[((size_t)b * 2048 + tok) * 512 + col] = acc[u][j][r] / lrun[u][r];
      }
}

// ---------------------------------------------------------------------------
extern "C" void kernel_launch(void* const* d_in, const int* in_sizes, int n_in,
                              void* d_out, int out_size, void* d_ws, size_t ws_size,
                              hipStream_t stream) {
  const float* x   = (const float*)d_in[0];
  const float* ang = (const float*)d_in[1];
  const float* anb = (const float*)d_in[2];
  const float* Wq  = (const float*)d_in[3];
  const float* bq  = (const float*)d_in[4];
  const float* Wk  = (const float*)d_in[5];
  const float* bk  = (const float*)d_in[6];
  const float* Wv  = (const float*)d_in[7];
  const float* bv  = (const float*)d_in[8];
  const float* lng = (const float*)d_in[9];
  const float* lnb = (const float*)d_in[10];
  const float* Wo  = (const float*)d_in[11];
  const float* bo  = (const float*)d_in[12];

  unsigned char* w = (unsigned char*)d_ws;
  size_t off = 0;
  auto alloc = [&](size_t bytes) {
    void* p = w + off;
    off += (bytes + 255) & ~(size_t)255;
    return p;
  };
  unsigned short* xn   = (unsigned short*)alloc(8192ull * 512 * 2);
  unsigned short* wtq  = (unsigned short*)alloc(512ull * 512 * 2);
  unsigned short* wtk  = (unsigned short*)alloc(256ull * 512 * 2);
  unsigned short* wtv  = (unsigned short*)alloc(256ull * 512 * 2);
  unsigned short* wto  = (unsigned short*)alloc(512ull * 512 * 2);
  unsigned short* qb   = (unsigned short*)alloc(8192ull * 512 * 2);
  unsigned short* kb   = (unsigned short*)alloc(8192ull * 256 * 2);
  unsigned short* vt   = (unsigned short*)alloc(8192ull * 256 * 2);
  float*          attn = (float*)alloc(8192ull * 512 * 4);
  unsigned short* xo   = (unsigned short*)alloc(8192ull * 512 * 2);

  // Weights -> bf16 transposed
  wconv_kernel<<<dim3(512 * 512 / 256), 256, 0, stream>>>(Wq, wtq, 512, 512 * 512);
  wconv_kernel<<<dim3(512 * 256 / 256), 256, 0, stream>>>(Wk, wtk, 256, 512 * 256);
  wconv_kernel<<<dim3(512 * 256 / 256), 256, 0, stream>>>(Wv, wtv, 256, 512 * 256);
  wconv_kernel<<<dim3(512 * 512 / 256), 256, 0, stream>>>(Wo, wto, 512, 512 * 512);

  // LN1 -> xn (bf16)
  ln_kernel<<<dim3(8192), 256, 0, stream>>>(x, ang, anb, xn);

  // QKV projections (WMMA), block tile 256x128
  gemm_bf16_kernel<<<dim3(32, 4), 256, 0, stream>>>(xn, wtq, bq, qb, 512, 1, 8);
  gemm_bf16_kernel<<<dim3(32, 2), 256, 0, stream>>>(xn, wtk, bk, kb, 256, 1, 4);
  gemm_bf16_kernel<<<dim3(32, 2), 256, 0, stream>>>(xn, wtv, bv, vt, 256, 3, 4);

  // Causal GQA attention with softmax_one (WMMA): 2048 blocks x 1 wave
  attn_kernel<<<dim3(2048), 32, 0, stream>>>(qb, kb, vt, attn);

  // LN2 -> xo (bf16), then output projection (WMMA) -> d_out (f32)
  ln_kernel<<<dim3(8192), 256, 0, stream>>>(attn, lng, lnb, xo);
  gemm_bf16_kernel<<<dim3(32, 4), 256, 0, stream>>>(xo, wto, bo, d_out, 512, 0, 0);
}
